// SpatioTemporalCentre_8177617732179
// MI455X (gfx1250) — compile-verified, matrix operand order checked by reading, not run on hardware
//
#include <hip/hip_runtime.h>
#include <hip/hip_bf16.h>
#include <math.h>

typedef __attribute__((ext_vector_type(16))) _Float16 v16h;
typedef __attribute__((ext_vector_type(8)))  _Float16 h8;
typedef __attribute__((ext_vector_type(8)))  float    v8f;
typedef _Float16 h16;

#define N_     64
#define C_     96
#define H_     18
#define W_     32
#define HW_    576
#define SIN_   320
#define HEADS_ 4
#define HD_    24
#define HID_   96
#define ENC_   320
#define M_     (N_*HW_)     /* 36864 */

// ---- WMMA helpers (CDNA5 16x16x32 f16, fp32 accum) ----
// A (16x32): lane&15 = row M; per-lane K = { off..off+7, off+16..off+23 }, off = (lane>>4)*8
// B (32x16): lane&15 = col N; same K striping.  C/D: row = r + 8*(lane>>4), col = lane&15.
__device__ __forceinline__ int fragK(int lane, int j) {
    return (((j >> 3) << 4) | ((lane >> 4) << 3) | (j & 7));
}
__device__ __forceinline__ v8f wmma16(v16h a, v16h b, v8f c) {
    return __builtin_amdgcn_wmma_f32_16x16x32_f16(false, a, false, b, (short)0, c, false, false);
}
__device__ __forceinline__ v8f zero8() {
    v8f z;
#pragma unroll
    for (int i = 0; i < 8; ++i) z[i] = 0.f;
    return z;
}
__device__ __forceinline__ h8 zeroh8() {
    h8 z;
#pragma unroll
    for (int i = 0; i < 8; ++i) z[i] = (h16)0.f;
    return z;
}
__device__ __forceinline__ v16h cat8(h8 lo, h8 hi) {
    return __builtin_shufflevector(lo, hi, 0,1,2,3,4,5,6,7,8,9,10,11,12,13,14,15);
}
__device__ __forceinline__ v16h ldA(const h16* rowbase, int kb, int lane) {
    const h16* p = rowbase + kb + ((lane >> 4) << 3);
    return cat8(*(const h8*)p, *(const h8*)(p + 16));
}
__device__ __forceinline__ v16h ldB(const h16* wtile, int lane) {
    const h16* p = wtile + (size_t)lane * 16;
    return cat8(*(const h8*)p, *(const h8*)(p + 8));
}
__device__ __forceinline__ float sigmoidf_(float x) { return 1.f / (1.f + __expf(-x)); }

// ---------------- K0: init enc output to -inf ----------------
__global__ __launch_bounds__(256) void k_init_enc(float* out_enc) {
    int t = blockIdx.x * blockDim.x + threadIdx.x;
    if (t < N_ * ENC_) out_enc[t] = -INFINITY;
}

// ---------------- K1: mod = x_state @ W_mod + b_mod ----------------
__global__ __launch_bounds__(256) void k_mod(const float* __restrict__ xs,
                                             const float* __restrict__ Wm,
                                             const float* __restrict__ bm,
                                             float* __restrict__ mod) {
    int t = blockIdx.x * blockDim.x + threadIdx.x;
    if (t >= N_ * 2 * C_) return;
    int n = t / (2 * C_), o = t % (2 * C_);
    float acc = bm[o];
    for (int s = 0; s < SIN_; ++s) acc += xs[n * SIN_ + s] * Wm[s * 2 * C_ + o];
    mod[t] = acc;
}

// ---------------- K2: pack weights to fragment-order f16 tiles ----------------
__global__ __launch_bounds__(256) void k_pack_w(const float* __restrict__ Wqkv,
                                                const float* __restrict__ Wproj,
                                                const float* __restrict__ Kzr,
                                                const float* __restrict__ Kc,
                                                const float* __restrict__ Wpos,
                                                h16* Wqkv16, h16* Wproj16,
                                                h16* Wzr16, h16* Wc16, h16* Wpos16) {
    int t = blockIdx.x * blockDim.x + threadIdx.x;
    int wt = t >> 9, lane = (t >> 4) & 31, j = t & 15;
    if (t < 3 * 18 * 512) {               // qkv: K<96, N=288
        int ks = wt / 18, nt = wt % 18;
        int K = ks * 32 + fragK(lane, j), col = nt * 16 + (lane & 15);
        Wqkv16[t] = (h16)Wqkv[K * 288 + col];
    }
    if (t < 3 * 6 * 512) {                // proj: K<96, N=96
        int ks = wt / 6, nt = wt % 6;
        int K = ks * 32 + fragK(lane, j), col = nt * 16 + (lane & 15);
        Wproj16[t] = (h16)Wproj[K * 96 + col];
    }
    if (t < 3 * 20 * 512) {               // pos: K<96, N=320; Wpos is [ENC][HID]
        int ks = wt / 20, nt = wt % 20;
        int K = ks * 32 + fragK(lane, j), col = nt * 16 + (lane & 15);
        Wpos16[t] = (h16)Wpos[col * 96 + K];
    }
    if (t < 54 * 12 * 512) {              // conv zr: K = (kh*3+kw)*192 + ci, N=192
        int ks = wt / 12, nt = wt % 12;
        int K = ks * 32 + fragK(lane, j), co = nt * 16 + (lane & 15);
        int kh3 = K / 192, ci = K - kh3 * 192, kh = kh3 / 3, kw = kh3 - kh * 3;
        Wzr16[t] = (h16)Kzr[((co * 192 + ci) * 3 + kh) * 3 + kw];
    }
    if (t < 54 * 6 * 512) {               // conv c
        int ks = wt / 6, nt = wt % 6;
        int K = ks * 32 + fragK(lane, j), co = nt * 16 + (lane & 15);
        int kh3 = K / 192, ci = K - kh3 * 192, kh = kh3 / 3, kw = kh3 - kh * 3;
        Wc16[t] = (h16)Kc[((co * 192 + ci) * 3 + kh) * 3 + kw];
    }
}

// ---------------- K3: FiLM + pack x to f16 [n][p][c]; grid(p, n), block c=96 ----------------
__global__ __launch_bounds__(96) void k_film(const float* __restrict__ x,
                                             const float* __restrict__ mod,
                                             h16* __restrict__ xf16) {
    int c = threadIdx.x, p = blockIdx.x, n = blockIdx.y;
    float mm = mod[n * 2 * C_ + c], mb = mod[n * 2 * C_ + C_ + c];
    xf16[((size_t)n * HW_ + p) * C_ + c] =
        (h16)(x[((size_t)n * C_ + c) * HW_ + p] * mm + mb);
}

// ---------------- K4: qkv GEMM; grid(9, 18, 64), 4 waves/WG ----------------
__global__ __launch_bounds__(128) void k_qkv(const h16* __restrict__ xf16,
                                             const h16* __restrict__ Wq,
                                             const float* __restrict__ bq,
                                             h16* __restrict__ q16,
                                             h16* __restrict__ k16,
                                             h16* __restrict__ v16t) {
    int lane = threadIdx.x & 31, wv = threadIdx.x >> 5;
    int n = blockIdx.z;
    int mt = blockIdx.x * 4 + wv;             // tile in image, 0..35
    int which = blockIdx.y / 6;               // scalar: 0=q 1=k 2=v
    int nloc  = blockIdx.y - which * 6;       // scalar: 0..5
    int pb0 = mt * 16;
    const h16* arow = xf16 + ((size_t)n * HW_ + pb0 + (lane & 15)) * 96;
    v8f acc = zero8();
    for (int ks = 0; ks < 3; ++ks) {
        v16h a = ldA(arow, ks * 32, lane);
        v16h b = ldB(Wq + (size_t)(ks * 18 + blockIdx.y) * 512, lane);
        acc = wmma16(a, b, acc);
    }
    int cc = nloc * 16 + (lane & 15);         // 0..95 within q/k/v
    float bias = bq[which * 96 + cc];
    int head = cc / HD_, dd = cc - head * HD_;
    int pb = pb0 + ((lane >> 4) << 3);
    h16* ptr; int stride;
    if (which == 0)      { ptr = q16  + (((size_t)n * HEADS_ + head) * HW_ + pb) * 32 + dd; stride = 32; }
    else if (which == 1) { ptr = k16  + (((size_t)n * HEADS_ + head) * HW_ + pb) * 32 + dd; stride = 32; }
    else                 { ptr = v16t + (((size_t)n * HEADS_ + head) * 32 + dd) * HW_ + pb; stride = 1; }
#pragma unroll
    for (int r = 0; r < 8; ++r) ptr[r * stride] = (h16)(acc[r] + bias);
}

// ---------------- K5: attention, one wave per (n, head, 16-row q tile) ----------------
__global__ __launch_bounds__(32) void k_attn(const h16* __restrict__ q16,
                                             const h16* __restrict__ k16,
                                             const h16* __restrict__ v16t,
                                             h16* __restrict__ o16) {
    __shared__ alignas(16) float    S[16 * HW_];
    __shared__ alignas(16) _Float16 P[16 * HW_];
    int lane = threadIdx.x;
    int mt = blockIdx.x, hh = blockIdx.y, n = blockIdx.z;
    const h16* qb  = q16  + ((size_t)n * HEADS_ + hh) * HW_ * 32;
    const h16* kb  = k16  + ((size_t)n * HEADS_ + hh) * HW_ * 32;
    const h16* vbt = v16t + ((size_t)n * HEADS_ + hh) * 32 * HW_;
    v16h aq = ldA(qb + (size_t)(mt * 16 + (lane & 15)) * 32, 0, lane);
    const float scale = 0.20412414523f; // 1/sqrt(24)
    for (int kt = 0; kt < 36; ++kt) {
        v16h bk = ldA(kb + (size_t)(kt * 16 + (lane & 15)) * 32, 0, lane);
        v8f s = zero8();
        s = wmma16(aq, bk, s);
#pragma unroll
        for (int r = 0; r < 8; ++r)
            S[(r + ((lane >> 4) << 3)) * HW_ + kt * 16 + (lane & 15)] = s[r] * scale;
    }
    __syncthreads();
    // softmax: lanes (row, half) each own 288 columns, vectorized float4 / h8
    int row = lane & 15, c0 = (lane >> 4) * 288;
    float4* S4 = (float4*)&S[row * HW_ + c0];
    float m = -INFINITY;
    for (int j = 0; j < 72; ++j) {
        float4 v = S4[j];
        m = fmaxf(m, fmaxf(fmaxf(v.x, v.y), fmaxf(v.z, v.w)));
    }
    m = fmaxf(m, __shfl_xor(m, 16, 32));
    float sum = 0.f;
    for (int j = 0; j < 72; ++j) {
        float4 v = S4[j];
        v.x = __expf(v.x - m); v.y = __expf(v.y - m);
        v.z = __expf(v.z - m); v.w = __expf(v.w - m);
        S4[j] = v;
        sum += (v.x + v.y) + (v.z + v.w);
    }
    sum += __shfl_xor(sum, 16, 32);
    float inv = 1.f / sum;
    h8* P8 = (h8*)&P[row * HW_ + c0];
    for (int j = 0; j < 36; ++j) {
        float4 a = S4[2 * j], b = S4[2 * j + 1];
        h8 o;
        o[0] = (h16)(a.x * inv); o[1] = (h16)(a.y * inv);
        o[2] = (h16)(a.z * inv); o[3] = (h16)(a.w * inv);
        o[4] = (h16)(b.x * inv); o[5] = (h16)(b.y * inv);
        o[6] = (h16)(b.z * inv); o[7] = (h16)(b.w * inv);
        P8[j] = o;
    }
    __syncthreads();
    // O = P @ V : K=576 (18 steps), N=32 (2 tiles, first 24 cols valid)
    v8f o0 = zero8(), o1 = zero8();
    for (int ks = 0; ks < 18; ++ks) {
        v16h ap  = ldA(&P[(lane & 15) * HW_], ks * 32, lane);
        v16h bv0 = ldA(vbt + (size_t)(lane & 15) * HW_, ks * 32, lane);
        v16h bv1 = ldA(vbt + (size_t)(16 + (lane & 15)) * HW_, ks * 32, lane);
        o0 = wmma16(ap, bv0, o0);
        o1 = wmma16(ap, bv1, o1);
    }
    int pb = mt * 16 + ((lane >> 4) << 3);
    h16* ob = o16 + ((size_t)n * HW_ + pb) * C_ + hh * HD_;
    int c1 = lane & 15, c2 = 16 + (lane & 15);
#pragma unroll
    for (int r = 0; r < 8; ++r) {
        ob[r * C_ + c1] = (h16)o0[r];
        if (c2 < HD_) ob[r * C_ + c2] = (h16)o1[r];
    }
}

// ---------------- K6: proj GEMM + residual; grid(9, 6, 64) ----------------
__global__ __launch_bounds__(128) void k_proj(const h16* __restrict__ o16,
                                              const h16* __restrict__ Wp,
                                              const float* __restrict__ bp,
                                              const float* __restrict__ x,
                                              h16* __restrict__ xh16,
                                              h16* __restrict__ xc16) {
    int lane = threadIdx.x & 31, wv = threadIdx.x >> 5;
    int n = blockIdx.z;
    int mt = blockIdx.x * 4 + wv;
    int pb0 = mt * 16;
    const h16* arow = o16 + ((size_t)n * HW_ + pb0 + (lane & 15)) * 96;
    v8f acc = zero8();
    for (int ks = 0; ks < 3; ++ks) {
        v16h a = ldA(arow, ks * 32, lane);
        v16h b = ldB(Wp + (size_t)(ks * 6 + blockIdx.y) * 512, lane);
        acc = wmma16(a, b, acc);
    }
    int col = blockIdx.y * 16 + (lane & 15);
    float bias = bp[col];
    int pb = pb0 + ((lane >> 4) << 3);
    const float* xr = x + ((size_t)n * C_ + col) * HW_ + pb;
    size_t di = ((size_t)n * HW_ + pb) * 192 + col;
#pragma unroll
    for (int r = 0; r < 8; ++r) {
        h16 y = (h16)(xr[r] + acc[r] + bias);
        xh16[di + (size_t)r * 192] = y;
        xc16[di + (size_t)r * 192] = y;
    }
}

// ---------------- K7: gather h -> xh16 channels 96..191; grid(p, n), block 96 ----------------
__global__ __launch_bounds__(96) void k_gather(const float* __restrict__ hid,
                                               const int* __restrict__ ak,
                                               h16* __restrict__ xh16) {
    int c = threadIdx.x, p = blockIdx.x, n = blockIdx.y;
    float h = hid[((size_t)ak[n] * HID_ + c) * HW_ + p];
    xh16[((size_t)n * HW_ + p) * 192 + 96 + c] = (h16)h;
}

// ---- conv 8-half run: one bounds check + one b128 load ----
__device__ __forceinline__ h8 conv_run(const h16* __restrict__ img, int K, int hh, int ww) {
    int kh3 = K / 192, ci = K - kh3 * 192;     // K = (kh*3+kw)*192 + ci
    int kh = kh3 / 3, kw = kh3 - kh * 3;
    int sh = hh + kh - 1, sw = ww + kw - 1;
    if (sh >= 0 && sh < H_ && sw >= 0 && sw < W_)
        return *(const h8*)(img + (size_t)(sh * W_ + sw) * 192 + ci);
    return zeroh8();
}

// ---------------- K8: gates conv 3x3, implicit GEMM, N=192; grid(9, 64) ----------------
__global__ __launch_bounds__(128) void k_convzr(const h16* __restrict__ xh16,
                                                const h16* __restrict__ Wzr,
                                                const float* __restrict__ bzr,
                                                float* __restrict__ zr32) {
    int lane = threadIdx.x & 31, wv = threadIdx.x >> 5;
    int n = blockIdx.y;
    int mt = blockIdx.x * 4 + wv;
    int pp = mt * 16 + (lane & 15), hh = pp >> 5, ww = pp & 31;
    const h16* img = xh16 + (size_t)n * HW_ * 192;
    v8f acc[12];
#pragma unroll
    for (int t = 0; t < 12; ++t) acc[t] = zero8();
    for (int ks = 0; ks < 54; ++ks) {
        if (ks < 53) __builtin_prefetch(Wzr + (size_t)((ks + 1) * 12) * 512, 0, 1);
        int K0 = ks * 32 + ((lane >> 4) << 3);
        v16h a = cat8(conv_run(img, K0, hh, ww), conv_run(img, K0 + 16, hh, ww));
#pragma unroll
        for (int nt = 0; nt < 12; ++nt) {
            v16h b = ldB(Wzr + (size_t)(ks * 12 + nt) * 512, lane);
            acc[nt] = wmma16(a, b, acc[nt]);
        }
    }
    size_t base = ((size_t)n * HW_ + mt * 16 + ((lane >> 4) << 3)) * 192;
#pragma unroll
    for (int nt = 0; nt < 12; ++nt) {
        int co = nt * 16 + (lane & 15);
        float bias = bzr[co];
#pragma unroll
        for (int r = 0; r < 8; ++r)
            zr32[base + (size_t)r * 192 + co] = acc[nt][r] + bias;   // [n][p][192]
    }
}

// ---------------- K9: r*h -> xc16 channels 96..191; grid(p, n), block 96 ----------------
__global__ __launch_bounds__(96) void k_gates(const float* __restrict__ zr32,
                                              const float* __restrict__ hid,
                                              const int* __restrict__ ak,
                                              h16* __restrict__ xc16) {
    int c = threadIdx.x, p = blockIdx.x, n = blockIdx.y;
    float r = sigmoidf_(zr32[((size_t)n * HW_ + p) * 192 + 96 + c]);
    float h = hid[((size_t)ak[n] * HID_ + c) * HW_ + p];
    xc16[((size_t)n * HW_ + p) * 192 + 96 + c] = (h16)(r * h);
}

// ---------------- K10: candidate conv 3x3 + fused GRU update; grid(9, 64) ----------------
__global__ __launch_bounds__(128) void k_convc(const h16* __restrict__ xc16,
                                               const h16* __restrict__ Wc,
                                               const float* __restrict__ bc,
                                               const float* __restrict__ zr32,
                                               const float* __restrict__ hid,
                                               const int* __restrict__ ak,
                                               float* __restrict__ hnew) {
    int lane = threadIdx.x & 31, wv = threadIdx.x >> 5;
    int n = blockIdx.y;
    int mt = blockIdx.x * 4 + wv;
    int pp = mt * 16 + (lane & 15), hh = pp >> 5, ww = pp & 31;
    const h16* img = xc16 + (size_t)n * HW_ * 192;
    v8f acc[6];
#pragma unroll
    for (int t = 0; t < 6; ++t) acc[t] = zero8();
    for (int ks = 0; ks < 54; ++ks) {
        if (ks < 53) __builtin_prefetch(Wc + (size_t)((ks + 1) * 6) * 512, 0, 1);
        int K0 = ks * 32 + ((lane >> 4) << 3);
        v16h a = cat8(conv_run(img, K0, hh, ww), conv_run(img, K0 + 16, hh, ww));
#pragma unroll
        for (int nt = 0; nt < 6; ++nt) {
            v16h b = ldB(Wc + (size_t)(ks * 6 + nt) * 512, lane);
            acc[nt] = wmma16(a, b, acc[nt]);
        }
    }
    int pb = mt * 16 + ((lane >> 4) << 3);
    int ha = ak[n];
    size_t zbase = ((size_t)n * HW_ + pb) * 192;
#pragma unroll
    for (int nt = 0; nt < 6; ++nt) {
        int co = nt * 16 + (lane & 15);
        float bias = bc[co];
        const float* hr = hid + ((size_t)ha * HID_ + co) * HW_ + pb;
        float* hw = hnew + ((size_t)n * HID_ + co) * HW_ + pb;
#pragma unroll
        for (int r = 0; r < 8; ++r) {
            float c = tanhf(acc[nt][r] + bias);
            float z = sigmoidf_(zr32[zbase + (size_t)r * 192 + co]);
            hw[r] = (1.f - z) * hr[r] + z * c;
        }
    }
}

// ---------------- K11a: 2x2 avg pool; grid(c, n), block 144 ----------------
__global__ __launch_bounds__(144) void k_pool(const float* __restrict__ hnew,
                                              float* __restrict__ out) {
    int t = threadIdx.x, c = blockIdx.x, n = blockIdx.y;
    int i = t >> 4, j = t & 15;
    const float* base = hnew + ((size_t)n * HID_ + c) * HW_ + (2 * i) * W_ + 2 * j;
    out[(((size_t)n * HID_ + c) * (H_ / 2) + i) * (W_ / 2) + j] =
        0.25f * ((base[0] + base[1]) + (base[W_] + base[W_ + 1]));
}

// ---------------- K11b: e = h_new + pos_emb, pack f16 [n][p][c]; grid(p, n), block 96 ----------------
__global__ __launch_bounds__(96) void k_packe(const float* __restrict__ hnew,
                                              const float* __restrict__ pos,
                                              h16* __restrict__ e16) {
    int c = threadIdx.x, p = blockIdx.x, n = blockIdx.y;
    float e = hnew[((size_t)n * HID_ + c) * HW_ + p] + pos[c * HW_ + p];
    e16[((size_t)n * HW_ + p) * HID_ + c] = (h16)e;
}

// ---------------- K12: enc GEMM + spatial max; grid(9, 20, 64) ----------------
__global__ __launch_bounds__(128) void k_enc(const h16* __restrict__ e16,
                                             const h16* __restrict__ Wpos,
                                             const float* __restrict__ bpos,
                                             float* __restrict__ out_enc) {
    int lane = threadIdx.x & 31, wv = threadIdx.x >> 5;
    int n = blockIdx.z;
    int mt = blockIdx.x * 4 + wv;
    const h16* arow = e16 + ((size_t)n * HW_ + mt * 16 + (lane & 15)) * 96;
    v8f acc = zero8();
    for (int ks = 0; ks < 3; ++ks) {
        v16h a = ldA(arow, ks * 32, lane);
        v16h b = ldB(Wpos + (size_t)(ks * 20 + blockIdx.y) * 512, lane);
        acc = wmma16(a, b, acc);
    }
    int col = blockIdx.y * 16 + (lane & 15);
    float m = -INFINITY;
#pragma unroll
    for (int r = 0; r < 8; ++r) m = fmaxf(m, acc[r]);
    m = fmaxf(m, __shfl_xor(m, 16, 32));
    if (lane < 16) atomicMax(&out_enc[n * ENC_ + col], m + bpos[col]);
}

// ---------------- host launch ----------------
extern "C" void kernel_launch(void* const* d_in, const int* in_sizes, int n_in,
                              void* d_out, int out_size, void* d_ws, size_t ws_size,
                              hipStream_t stream) {
    const float* x      = (const float*)d_in[0];
    const float* xs     = (const float*)d_in[1];
    const int*   ak     = (const int*)d_in[2];
    const float* hid    = (const float*)d_in[3];
    const float* W_mod  = (const float*)d_in[4];
    const float* b_mod  = (const float*)d_in[5];
    const float* W_qkv  = (const float*)d_in[6];
    const float* b_qkv  = (const float*)d_in[7];
    const float* W_proj = (const float*)d_in[8];
    const float* b_proj = (const float*)d_in[9];
    const float* K_zr   = (const float*)d_in[10];
    const float* b_zr   = (const float*)d_in[11];
    const float* K_c    = (const float*)d_in[12];
    const float* b_c    = (const float*)d_in[13];
    const float* pos    = (const float*)d_in[14];
    const float* W_pos  = (const float*)d_in[15];
    const float* b_pos  = (const float*)d_in[16];

    float* out_pool = (float*)d_out;                    // [64,96,9,16]
    float* out_enc  = out_pool + (size_t)N_ * HID_ * (H_/2) * (W_/2);

    // workspace carve-up (256B aligned); reuse: xf16->o16, q16->e16, xh16->hnew
    char* wsp = (char*)d_ws;
    auto carve = [&](size_t bytes) { char* p = wsp; wsp += (bytes + 255) & ~(size_t)255; return p; };
    float* mod     = (float*)carve((size_t)N_ * 2 * C_ * 4);
    h16*   Wqkv16  = (h16*)carve((size_t)3 * 18 * 512 * 2);
    h16*   Wproj16 = (h16*)carve((size_t)3 * 6 * 512 * 2);
    h16*   Wzr16   = (h16*)carve((size_t)54 * 12 * 512 * 2);
    h16*   Wc16    = (h16*)carve((size_t)54 * 6 * 512 * 2);
    h16*   Wpos16  = (h16*)carve((size_t)3 * 20 * 512 * 2);
    h16*   buf1    = (h16*)carve((size_t)M_ * 96 * 2);                 // xf16, then o16
    h16*   q16     = (h16*)carve((size_t)N_ * HEADS_ * HW_ * 32 * 2);  // then e16
    h16*   kv16    = (h16*)carve((size_t)2 * N_ * HEADS_ * HW_ * 32 * 2);
    h16*   xh16    = (h16*)carve((size_t)N_ * HW_ * 192 * 2);          // then hnew (same bytes)
    h16*   xc16    = (h16*)carve((size_t)N_ * HW_ * 192 * 2);
    float* zr32    = (float*)carve((size_t)N_ * HW_ * 192 * 4);
    h16* k16  = kv16;
    h16* v16t = kv16 + (size_t)N_ * HEADS_ * HW_ * 32;
    h16* xf16 = buf1;
    h16* o16  = buf1;
    h16* e16  = q16;
    float* hnew = (float*)xh16;

    k_init_enc<<<(N_ * ENC_ + 255) / 256, 256, 0, stream>>>(out_enc);
    k_pack_w<<<(54 * 12 * 512 + 255) / 256, 256, 0, stream>>>(W_qkv, W_proj, K_zr, K_c, W_pos,
                                                              Wqkv16, Wproj16, Wzr16, Wc16, Wpos16);
    k_mod<<<(N_ * 2 * C_ + 255) / 256, 256, 0, stream>>>(xs, W_mod, b_mod, mod);
    hipMemsetAsync(q16, 0, (size_t)N_ * HEADS_ * HW_ * 32 * 2, stream);
    hipMemsetAsync(kv16, 0, (size_t)2 * N_ * HEADS_ * HW_ * 32 * 2, stream);
    k_film<<<dim3(HW_, N_), 96, 0, stream>>>(x, mod, xf16);
    k_qkv<<<dim3(9, 18, N_), 128, 0, stream>>>(xf16, Wqkv16, b_qkv, q16, k16, v16t);
    k_attn<<<dim3(HW_ / 16, HEADS_, N_), 32, 0, stream>>>(q16, k16, v16t, o16);
    k_proj<<<dim3(9, 6, N_), 128, 0, stream>>>(o16, Wproj16, b_proj, x, xh16, xc16);
    k_gather<<<dim3(HW_, N_), 96, 0, stream>>>(hid, ak, xh16);
    k_convzr<<<dim3(9, N_), 128, 0, stream>>>(xh16, Wzr16, b_zr, zr32);
    k_gates<<<dim3(HW_, N_), 96, 0, stream>>>(zr32, hid, ak, xc16);
    k_convc<<<dim3(9, N_), 128, 0, stream>>>(xc16, Wc16, b_c, zr32, hid, ak, hnew);
    k_pool<<<dim3(HID_, N_), 144, 0, stream>>>(hnew, out_pool);
    k_packe<<<dim3(HW_, N_), 96, 0, stream>>>(hnew, pos, e16);
    k_enc<<<dim3(9, 20, N_), 128, 0, stream>>>(e16, Wpos16, b_pos, out_enc);
}
